// MultiHeadedAttention_47880295415993
// MI455X (gfx1250) — matrix operand order repaired and checked by reading, not compiled, not run on hardware
//
#include <hip/hip_runtime.h>

// ---------------------------------------------------------------------------
// MultiHeadedAttention for MI455X (gfx1250, wave32, WMMA bf16 16x16x32)
//   B=4, S=2048, D=1024, H=16, DK=64
// Pipeline:
//   1) f32 -> bf16 casts (X, Wq, Wk, Wv, Wo)
//   2) Q/K/V = X @ W^T  (WMMA NT GEMM, async-to-LDS double-buffered,
//      bf16 in / f32 acc / bf16 out, written [B][H][S][DK])
//   3) flash attention per (b, h, 64-row q tile): QK^T, online softmax, P·V
//      (V staged TRANSPOSED in LDS so all fragments are b128 loads)
//   4) out = ctx @ Wo^T (WMMA NT GEMM, f32 out)
// ---------------------------------------------------------------------------

typedef unsigned short u16;
typedef unsigned int   u32;

typedef __attribute__((ext_vector_type(16))) __bf16 v16bf;
typedef __attribute__((ext_vector_type(8)))  float  v8f;
typedef int v4i __attribute__((vector_size(16)));   // matches builtin param type

#define GAS __attribute__((address_space(1)))
#define LAS __attribute__((address_space(3)))

#if defined(__has_builtin)
#if __has_builtin(__builtin_amdgcn_global_load_async_to_lds_b128) && \
    __has_builtin(__builtin_amdgcn_s_wait_asynccnt)
#define USE_ASYNC 1
#endif
#endif
#ifndef USE_ASYNC
#define USE_ASYNC 0
#endif

union BF16Frag {
    v16bf v;
    u16   h[16];
    u32   w[8];
    uint4 q[2];
};

union U16x32 {
    uint4 q[4];
    u16   h[32];
};

__device__ __forceinline__ u16 f2bf(float f) {
    u32 u = __builtin_bit_cast(u32, f);
    u32 r = u + 0x7FFFu + ((u >> 16) & 1u);   // round-to-nearest-even
    return (u16)(r >> 16);
}

#if USE_ASYNC
// one 16B global -> LDS async copy (ASYNCcnt-tracked, no VGPR data)
__device__ __forceinline__ void async_cp16(const u16* g, u16* l) {
    __builtin_amdgcn_global_load_async_to_lds_b128(
        (GAS v4i*)g, (LAS v4i*)l, 0, 0);
}
#endif

// --------------------------- f32 -> bf16 cast ------------------------------
__global__ __launch_bounds__(256)
void cvt_f32_to_bf16(const float* __restrict__ in, u16* __restrict__ out, int n) {
    int i = (blockIdx.x * 256 + threadIdx.x) * 4;
    if (i >= n) return;
    float4 v = *(const float4*)(in + i);
    uint2 o;
    o.x = (u32)f2bf(v.x) | ((u32)f2bf(v.y) << 16);
    o.y = (u32)f2bf(v.z) | ((u32)f2bf(v.w) << 16);
    *(uint2*)(out + i) = o;
}

// --------------------------- NT GEMM (WMMA) --------------------------------
// C[m,n] = sum_k A[m,k] * W[n,k];  M=8192, N=K=1024.
// Block tile 128x128 (256 thr = 8 waves, wave tile 32x64), K-step 32,
// double-buffered LDS with async-to-LDS copies when available.
// MODE 0: f32 out, row-major [M,1024]     (final projection -> d_out)
// MODE 1: bf16 out, permuted [B][H][S][DK] (Q/K/V for attention)
template <int MODE>
__global__ __launch_bounds__(256)
void gemm_nt_bf16(const u16* __restrict__ A,
                  const u16* __restrict__ W,
                  float* __restrict__ outF,
                  u16*  __restrict__ outP) {
    __shared__ u16 As[2][128 * 40];   // rows padded to 40 (16B-aligned rows)
    __shared__ u16 Ws[2][128 * 40];

    const int tid  = threadIdx.x;
    const int lane = tid & 31;
    const int wv   = tid >> 5;
    const int hl   = lane >> 4;     // half-wave (0/1)
    const int l16  = lane & 15;

    const int m0 = blockIdx.y * 128;
    const int n0 = blockIdx.x * 128;
    const int wm = (wv & 3) * 32;   // wave M offset
    const int wn = (wv >> 2) * 64;  // wave N offset

    v8f acc[2][4];
#pragma unroll
    for (int i = 0; i < 2; ++i)
#pragma unroll
        for (int j = 0; j < 4; ++j)
#pragma unroll
            for (int e = 0; e < 8; ++e) acc[i][j][e] = 0.0f;

    const int lrow = tid >> 1;        // 0..127
    const int lcol = (tid & 1) * 16;  // 0 / 16
    const int loff = lrow * 40 + lcol;
    const u16* gA = A + (size_t)(m0 + lrow) * 1024 + lcol;
    const u16* gW = W + (size_t)(n0 + lrow) * 1024 + lcol;

#if USE_ASYNC
    // prologue: stage K-tile 0 into buffer 0
    async_cp16(gA,     &As[0][loff]);
    async_cp16(gA + 8, &As[0][loff + 8]);
    async_cp16(gW,     &Ws[0][loff]);
    async_cp16(gW + 8, &Ws[0][loff + 8]);
    __builtin_amdgcn_s_wait_asynccnt(0);
    __syncthreads();
#endif

    for (int k0 = 0; k0 < 1024; k0 += 32) {
#if USE_ASYNC
        const int cur = (k0 >> 5) & 1;
        const int nxt = cur ^ 1;
        if (k0 + 32 < 1024) {       // prefetch next tile while computing
            async_cp16(gA + k0 + 32, &As[nxt][loff]);
            async_cp16(gA + k0 + 40, &As[nxt][loff + 8]);
            async_cp16(gW + k0 + 32, &Ws[nxt][loff]);
            async_cp16(gW + k0 + 40, &Ws[nxt][loff + 8]);
        }
#else
        const int cur = 0;
        uint4 a0 = *(const uint4*)(gA + k0);
        uint4 a1 = *(const uint4*)(gA + k0 + 8);
        uint4 w0 = *(const uint4*)(gW + k0);
        uint4 w1 = *(const uint4*)(gW + k0 + 8);
        if (k0 + 32 < 1024) {
            __builtin_prefetch(gA + k0 + 32, 0, 0);
            __builtin_prefetch(gW + k0 + 32, 0, 0);
        }
        *(uint4*)&As[0][loff]     = a0;
        *(uint4*)&As[0][loff + 8] = a1;
        *(uint4*)&Ws[0][loff]     = w0;
        *(uint4*)&Ws[0][loff + 8] = w1;
        __syncthreads();
#endif

        // A fragments (16x32 bf16): lane = M row, 16 contiguous K per half
        BF16Frag af[2];
#pragma unroll
        for (int mi = 0; mi < 2; ++mi) {
            const int row = wm + mi * 16 + l16;
#pragma unroll
            for (int r = 0; r < 8; ++r) {
                const int kk = 2 * r + hl * 16;
                af[mi].w[r] = *(const u32*)&As[cur][row * 40 + kk];
            }
        }
        // B fragments (32x16 bf16): B[k][n] = W[n][k] -> lane n reads 16 contig k
        BF16Frag bfr[4];
#pragma unroll
        for (int ni = 0; ni < 4; ++ni) {
            const int row = wn + ni * 16 + l16;
            bfr[ni].q[0] = *(const uint4*)&Ws[cur][row * 40 + hl * 16];
            bfr[ni].q[1] = *(const uint4*)&Ws[cur][row * 40 + hl * 16 + 8];
        }
#pragma unroll
        for (int mi = 0; mi < 2; ++mi)
#pragma unroll
            for (int ni = 0; ni < 4; ++ni)
                acc[mi][ni] = __builtin_amdgcn_wmma_f32_16x16x32_bf16(
                    false, af[mi].v, false, bfr[ni].v,
                    (short)0, acc[mi][ni], false, false);

#if USE_ASYNC
        __builtin_amdgcn_s_wait_asynccnt(0);   // next tile landed
#endif
        __syncthreads();
    }

    // Epilogue: C/D layout -> lane l16 = N col, VGPR r + hl*8 = M row
#pragma unroll
    for (int mi = 0; mi < 2; ++mi)
#pragma unroll
        for (int r = 0; r < 8; ++r) {
            const int gm = m0 + wm + mi * 16 + r + hl * 8;
#pragma unroll
            for (int ni = 0; ni < 4; ++ni) {
                const int gn = n0 + wn + ni * 16 + l16;
                const float val = acc[mi][ni][r];
                if (MODE == 0) {
                    outF[(size_t)gm * 1024 + gn] = val;
                } else {
                    const int bb = gm >> 11, ss = gm & 2047;  // m = b*S + s
                    const int hh = gn >> 6,  dk = gn & 63;    // n = h*DK + dk
                    outP[(((size_t)(bb * 16 + hh) * 2048) + ss) * 64 + dk] = f2bf(val);
                }
            }
        }
}

// --------------------------- Flash attention -------------------------------
// grid = (S/64, H, B), block = 128 (4 waves); each wave owns 16 q rows.
// K staged row-major (async when available); V staged TRANSPOSED so that
// both K and V WMMA B-fragments are aligned ds_load_b128s.
__global__ __launch_bounds__(128)
void flash_attn(const u16* __restrict__ Qb, const u16* __restrict__ Kb,
                const u16* __restrict__ Vb, u16* __restrict__ ctxb) {
    __shared__ u16 Kl[64 * 72];      // [keypos][dk], padded rows (16B aligned)
    __shared__ u16 Vt[64 * 72];      // [dk][keypos]  (transposed at staging)
    __shared__ u16 Pl[4][16 * 72];   // per-wave P tile [m][keypos]

    const int qt = blockIdx.x, h = blockIdx.y, b = blockIdx.z;
    const int tid  = threadIdx.x;
    const int lane = tid & 31;
    const int wv   = tid >> 5;
    const int hl   = lane >> 4;
    const int l16  = lane & 15;
    const size_t bh = (size_t)(b * 16 + h) * 2048 * 64;
    const float scale = 0.125f;      // 1/sqrt(64)

    // Q A-fragments: 16 rows x 64 k = 2 K-steps, straight from global (bf16)
    BF16Frag aq[2];
    const u16* Qp = Qb + bh + (size_t)(qt * 64 + wv * 16 + l16) * 64;
#pragma unroll
    for (int ks = 0; ks < 2; ++ks)
#pragma unroll
        for (int r = 0; r < 8; ++r) {
            const int kk = ks * 32 + 2 * r + hl * 16;
            aq[ks].w[r] = *(const u32*)(Qp + kk);
        }

    v8f o[4];
#pragma unroll
    for (int d = 0; d < 4; ++d)
#pragma unroll
        for (int e = 0; e < 8; ++e) o[d][e] = 0.0f;
    float m_i[8], l_i[8];
#pragma unroll
    for (int r = 0; r < 8; ++r) { m_i[r] = -1e30f; l_i[r] = 0.0f; }

    const int lrow = tid >> 1;        // 0..63 : key row this thread stages
    const int lcol = (tid & 1) * 32;  // 0 / 32: dk start

    for (int kb = 0; kb < 2048; kb += 64) {
        const u16* gK = Kb + bh + (size_t)(kb + lrow) * 64 + lcol;
        const u16* gV = Vb + bh + (size_t)(kb + lrow) * 64 + lcol;

        // K: row-major stage (async path touches no VGPRs)
#if USE_ASYNC
#pragma unroll
        for (int j = 0; j < 4; ++j)
            async_cp16(gK + j * 8, &Kl[lrow * 72 + lcol + j * 8]);
#else
#pragma unroll
        for (int j = 0; j < 4; ++j) {
            uint4 kv = *(const uint4*)(gK + j * 8);
            *(uint4*)&Kl[lrow * 72 + lcol + j * 8] = kv;
        }
#endif
        // V: load row, scatter transposed -> Vt[dk][keypos]
        U16x32 vr;
#pragma unroll
        for (int j = 0; j < 4; ++j) vr.q[j] = *(const uint4*)(gV + j * 8);
#pragma unroll
        for (int j = 0; j < 32; ++j)
            Vt[(lcol + j) * 72 + lrow] = vr.h[j];

#if USE_ASYNC
        __builtin_amdgcn_s_wait_asynccnt(0);
#endif
        __syncthreads();

        // S = Q K^T over this key block (4 N-tiles of 16 keys)
        v8f s[4];
#pragma unroll
        for (int nt = 0; nt < 4; ++nt) {
            BF16Frag bk0, bk1;
            const int krow = (nt * 16 + l16) * 72;
            bk0.q[0] = *(const uint4*)&Kl[krow + hl * 16];
            bk0.q[1] = *(const uint4*)&Kl[krow + hl * 16 + 8];
            bk1.q[0] = *(const uint4*)&Kl[krow + 32 + hl * 16];
            bk1.q[1] = *(const uint4*)&Kl[krow + 32 + hl * 16 + 8];
            v8f zz;
#pragma unroll
            for (int e = 0; e < 8; ++e) zz[e] = 0.0f;
            s[nt] = __builtin_amdgcn_wmma_f32_16x16x32_bf16(
                false, aq[0].v, false, bk0.v, (short)0, zz, false, false);
            s[nt] = __builtin_amdgcn_wmma_f32_16x16x32_bf16(
                false, aq[1].v, false, bk1.v, (short)0, s[nt], false, false);
        }

        // online softmax (row = r + hl*8; row spans lanes 0..15 of this half)
#pragma unroll
        for (int r = 0; r < 8; ++r) {
            float sv[4];
            float mx = -1e30f;
#pragma unroll
            for (int nt = 0; nt < 4; ++nt) {
                sv[nt] = s[nt][r] * scale;
                mx = fmaxf(mx, sv[nt]);
            }
#pragma unroll
            for (int d = 1; d < 16; d <<= 1)
                mx = fmaxf(mx, __shfl_xor(mx, d, 32));   // stays inside half
            const float mnew  = fmaxf(m_i[r], mx);
            const float alpha = __expf(m_i[r] - mnew);
            float rowsum = 0.0f;
#pragma unroll
            for (int nt = 0; nt < 4; ++nt) {
                const float p = __expf(sv[nt] - mnew);
                rowsum += p;
                Pl[wv][(r + hl * 8) * 72 + nt * 16 + l16] = f2bf(p);
            }
#pragma unroll
            for (int d = 1; d < 16; d <<= 1)
                rowsum += __shfl_xor(rowsum, d, 32);
            l_i[r] = l_i[r] * alpha + rowsum;
            m_i[r] = mnew;
#pragma unroll
            for (int dt = 0; dt < 4; ++dt) o[dt][r] *= alpha;
        }

        // P @ V : A-fragments from per-wave Pl; V B-fragments from Vt (b128)
        BF16Frag ap[2];
#pragma unroll
        for (int ks = 0; ks < 2; ++ks)
#pragma unroll
            for (int r = 0; r < 8; ++r) {
                const int kk = ks * 32 + 2 * r + hl * 16;
                ap[ks].w[r] = *(const u32*)&Pl[wv][l16 * 72 + kk];
            }
#pragma unroll
        for (int dt = 0; dt < 4; ++dt) {
            BF16Frag bv0, bv1;
            const int vrow = (dt * 16 + l16) * 72;
            bv0.q[0] = *(const uint4*)&Vt[vrow + hl * 16];
            bv0.q[1] = *(const uint4*)&Vt[vrow + hl * 16 + 8];
            bv1.q[0] = *(const uint4*)&Vt[vrow + 32 + hl * 16];
            bv1.q[1] = *(const uint4*)&Vt[vrow + 32 + hl * 16 + 8];
            o[dt] = __builtin_amdgcn_wmma_f32_16x16x32_bf16(
                false, ap[0].v, false, bv0.v, (short)0, o[dt], false, false);
            o[dt] = __builtin_amdgcn_wmma_f32_16x16x32_bf16(
                false, ap[1].v, false, bv1.v, (short)0, o[dt], false, false);
        }
        __syncthreads();   // before next block overwrites Kl/Vt
    }

    // epilogue: ctx[b, q, h*64+dk] in bf16, row-major [B*S, 1024]
#pragma unroll
    for (int r = 0; r < 8; ++r) {
        const float inv = 1.0f / l_i[r];
        const int qg = qt * 64 + wv * 16 + r + hl * 8;
#pragma unroll
        for (int dt = 0; dt < 4; ++dt) {
            const int dk = dt * 16 + l16;
            ctxb[(size_t)(b * 2048 + qg) * 1024 + h * 64 + dk] = f2bf(o[dt][r] * inv);
        }
    }
}

// ----------------------------- launch --------------------------------------
extern "C" void kernel_launch(void* const* d_in, const int* in_sizes, int n_in,
                              void* d_out, int out_size, void* d_ws, size_t ws_size,
                              hipStream_t stream) {
    const float* X  = (const float*)d_in[0];
    const float* Wq = (const float*)d_in[1];
    const float* Wk = (const float*)d_in[2];
    const float* Wv = (const float*)d_in[3];
    const float* Wo = (const float*)d_in[4];

    char* ws = (char*)d_ws;
    const size_t SZ_X = (size_t)8192 * 1024 * 2;  // 16 MiB bf16
    const size_t SZ_W = (size_t)1024 * 1024 * 2;  //  2 MiB bf16
    u16* Xb  = (u16*)(ws);
    u16* Wqb = (u16*)(ws + SZ_X);
    u16* Wkb = (u16*)(ws + SZ_X + SZ_W);
    u16* Wvb = (u16*)(ws + SZ_X + 2 * SZ_W);
    u16* Wob = (u16*)(ws + SZ_X + 3 * SZ_W);
    u16* Qb  = (u16*)(ws + SZ_X + 4 * SZ_W);
    u16* Kb  = (u16*)(ws + 2 * SZ_X + 4 * SZ_W);
    u16* Vb  = (u16*)(ws + 3 * SZ_X + 4 * SZ_W);
    u16* Cb  = (u16*)(ws + 4 * SZ_X + 4 * SZ_W);

    // 1) casts
    cvt_f32_to_bf16<<<8192 * 1024 / 1024, 256, 0, stream>>>(X,  Xb,  8192 * 1024);
    cvt_f32_to_bf16<<<1024 * 1024 / 1024, 256, 0, stream>>>(Wq, Wqb, 1024 * 1024);
    cvt_f32_to_bf16<<<1024 * 1024 / 1024, 256, 0, stream>>>(Wk, Wkb, 1024 * 1024);
    cvt_f32_to_bf16<<<1024 * 1024 / 1024, 256, 0, stream>>>(Wv, Wvb, 1024 * 1024);
    cvt_f32_to_bf16<<<1024 * 1024 / 1024, 256, 0, stream>>>(Wo, Wob, 1024 * 1024);

    // 2) Q/K/V projections -> [B][H][S][DK] bf16
    dim3 ggrid(8, 64);
    gemm_nt_bf16<1><<<ggrid, 256, 0, stream>>>(Xb, Wqb, nullptr, Qb);
    gemm_nt_bf16<1><<<ggrid, 256, 0, stream>>>(Xb, Wkb, nullptr, Kb);
    gemm_nt_bf16<1><<<ggrid, 256, 0, stream>>>(Xb, Wvb, nullptr, Vb);

    // 3) flash attention -> ctx bf16 [B*S, 1024]
    flash_attn<<<dim3(32, 16, 4), 128, 0, stream>>>(Qb, Kb, Vb, Cb);

    // 4) output projection -> f32 d_out
    gemm_nt_bf16<0><<<ggrid, 256, 0, stream>>>(Cb, Wob, (float*)d_out, nullptr);
}
